// Attention_4776003633693
// MI455X (gfx1250) — matrix operand order repaired and checked by reading, not compiled, not run on hardware
//
#include <hip/hip_runtime.h>

typedef __attribute__((ext_vector_type(16))) _Float16 v16h;
typedef __attribute__((ext_vector_type(8)))  _Float16 v8h;
typedef __attribute__((ext_vector_type(8)))  float    v8f;
typedef __attribute__((ext_vector_type(4)))  int      v4i;

#define BN   16
#define CC   512
#define NN   1024
#define CPG  16          // channels per group (512 / 32 groups)
#define EPSV 1e-5f

#if defined(__has_builtin)
#if __has_builtin(__builtin_amdgcn_global_load_async_to_lds_b128)
#define USE_ASYNC_LDS 1
#endif
#endif
#ifndef USE_ASYNC_LDS
#define USE_ASYNC_LDS 0
#endif

typedef __attribute__((address_space(1))) v4i as1_v4i;
typedef __attribute__((address_space(3))) v4i as3_v4i;

// ---------------------------------------------------------------------------
// Stage a contiguous global region into LDS with a 256-thread block.
// Uses GLOBAL_LOAD_ASYNC_TO_LDS_B128 (ASYNCcnt path) when the builtin exists.
// bytes must be a multiple of 4096. Caller must __syncthreads() afterwards.
// ---------------------------------------------------------------------------
__device__ __forceinline__ void stage_to_lds(const _Float16* __restrict__ g,
                                             _Float16* __restrict__ l,
                                             int bytes, int tid) {
#if USE_ASYNC_LDS
  for (int off = tid * 16; off < bytes; off += 256 * 16) {
    __builtin_amdgcn_global_load_async_to_lds_b128(
        (as1_v4i*)(size_t)((const char*)g + off),
        (as3_v4i*)(unsigned)(size_t)((char*)l + off),
        0, 0);
  }
#if __has_builtin(__builtin_amdgcn_s_wait_asynccnt)
  __builtin_amdgcn_s_wait_asynccnt(0);
#else
  asm volatile("s_wait_asynccnt 0" ::: "memory");
#endif
#else
  for (int off = tid * 16; off < bytes; off += 256 * 16) {
    *(v8h*)((char*)l + off) = *(const v8h*)((const char*)g + off);
  }
#endif
}

// ---------------------------------------------------------------------------
// WMMA operand loaders (CDNA5 wave32 layouts, CDNA5 ISA §7.12.2)
// ---------------------------------------------------------------------------

// A-matrix 16x32 f16: lane L -> row m = L&15, K-half kb = 8*(L>>4);
// VGPR0..3 hold K=kb..kb+7, VGPR4..7 hold K=16+kb..16+kb+7.
__device__ __forceinline__ v16h load_a16x32(const _Float16* base, int ld, int lane) {
  const int m  = lane & 15;
  const int kb = (lane >> 4) << 3;
  const _Float16* p = base + (size_t)m * ld + kb;
  v8h lo = *(const v8h*)(p);
  v8h hi = *(const v8h*)(p + 16);
  return __builtin_shufflevector(lo, hi, 0,1,2,3,4,5,6,7,8,9,10,11,12,13,14,15);
}

// B-matrix 32x16 f16 stored K-contiguous (i.e. B^T row-major, row = output col n):
// lane L -> col n = L&15, kb = 8*(L>>4); elems 0..7 hold K=kb..kb+7, 8..15 hold K=16+kb..16+kb+7.
__device__ __forceinline__ v16h load_b32x16(const _Float16* baseT, int ld, int lane) {
  const int n  = lane & 15;
  const int k0 = (lane >> 4) << 3;
  const _Float16* p = baseT + (size_t)n * ld + k0;
  v8h lo = *(const v8h*)(p);
  v8h hi = *(const v8h*)(p + 16);
  return __builtin_shufflevector(lo, hi, 0,1,2,3,4,5,6,7,8,9,10,11,12,13,14,15);
}

__device__ __forceinline__ v8f wmma_f16(v16h a, v16h b, v8f c) {
  return __builtin_amdgcn_wmma_f32_16x16x32_f16(false, a, false, b,
                                                (short)0, c, false, false);
}

// f32 C/D 16x16: VGPR r, lane L -> m = r + 8*(L>>4), n = L&15.

// ---------------------------------------------------------------------------
// 1) GroupNorm -> f16 activations in [b][n][c] layout (c contiguous)
// ---------------------------------------------------------------------------
__global__ void gn_kernel(const float* __restrict__ x, const float* __restrict__ sc,
                          const float* __restrict__ bi, _Float16* __restrict__ xn16) {
  __shared__ float s1[256];
  __shared__ float s2[256];
  const int b   = blockIdx.x >> 5;   // 32 groups
  const int g   = blockIdx.x & 31;
  const int tid = threadIdx.x;
  const float* xg = x + ((size_t)b * CC + g * CPG) * NN;

  float sum = 0.f, sq = 0.f;
  for (int i = tid; i < CPG * NN; i += 256) {
    float v = xg[i];
    sum += v; sq += v * v;
  }
  s1[tid] = sum; s2[tid] = sq;
  __syncthreads();
  for (int s = 128; s > 0; s >>= 1) {
    if (tid < s) { s1[tid] += s1[tid + s]; s2[tid] += s2[tid + s]; }
    __syncthreads();
  }
  const float inv_n = 1.f / (float)(CPG * NN);
  const float mean  = s1[0] * inv_n;
  const float var   = s2[0] * inv_n - mean * mean;
  const float istd  = rsqrtf(var + EPSV);

  for (int i = tid; i < CPG * NN; i += 256) {
    const int cl = i >> 10;             // i / NN
    const int n  = i & (NN - 1);
    const int c  = g * CPG + cl;
    const float v = (xg[i] - mean) * istd * sc[c] + bi[c];
    xn16[((size_t)b * NN + n) * CC + c] = (_Float16)v;
  }
}

// ---------------------------------------------------------------------------
// 2) Convert the four CxC weight matrices to f16 (packed wq|wk|wv|wp)
// ---------------------------------------------------------------------------
__global__ void wcvt_kernel(const float* __restrict__ wq, const float* __restrict__ wk,
                            const float* __restrict__ wv, const float* __restrict__ wp,
                            _Float16* __restrict__ w16) {
  const int idx   = blockIdx.x * 256 + threadIdx.x;    // 4*CC*CC total
  const int sz    = CC * CC;
  const int which = idx / sz;
  const int off   = idx - which * sz;
  const float* src = which == 0 ? wq : which == 1 ? wk : which == 2 ? wv : wp;
  w16[idx] = (_Float16)src[off];
}

// ---------------------------------------------------------------------------
// 3) QKV projections: one block per (b, 16-wide n-tile). The shared xn tile
//    (16 x 512 f16 = 16KB) is staged to LDS once; 96 output tiles
//    (3 matrices x 32 cout-tiles) are computed 2-at-a-time per wave so each
//    LDS B-load feeds two WMMAs. q,k stored transposed [b][n][c]; v [b][c][n].
// ---------------------------------------------------------------------------
__device__ __forceinline__ void qkv_store(int mat, int c0, v8f acc,
                                          const float* bq, const float* bk,
                                          const float* bv,
                                          _Float16* q16, _Float16* k16,
                                          _Float16* v16,
                                          int b, int n0, int mrow, int ncol) {
  const float* bias = (mat == 0) ? bq : (mat == 1) ? bk : bv;
  if (mat < 2) {  // q/k: transposed store, 8 contiguous halfs per lane
    _Float16* dst = (mat == 0 ? q16 : k16)
                  + ((size_t)b * NN + n0 + ncol) * CC + c0 + mrow;
    v8h pk;
#pragma unroll
    for (int r = 0; r < 8; ++r) pk[r] = (_Float16)(acc[r] + bias[c0 + mrow + r]);
    *(v8h*)dst = pk;
  } else {        // v: [b][c][n]
#pragma unroll
    for (int r = 0; r < 8; ++r) {
      const int c = c0 + mrow + r;
      v16[((size_t)b * CC + c) * NN + n0 + ncol] = (_Float16)(acc[r] + bias[c]);
    }
  }
}

__global__ void qkv_kernel(const _Float16* __restrict__ xn16,
                           const _Float16* __restrict__ w16,
                           const float* __restrict__ bq, const float* __restrict__ bk,
                           const float* __restrict__ bv,
                           _Float16* __restrict__ q16, _Float16* __restrict__ k16,
                           _Float16* __restrict__ v16) {
  __shared__ _Float16 xs[16 * CC];        // 16 KB activation tile [n_local][c]
  const int tid  = threadIdx.x;
  const int lane = tid & 31;
  const int wave = tid >> 5;
  const int b  = blockIdx.x >> 6;
  const int nt = blockIdx.x & 63;
  const int n0 = nt * 16;

  stage_to_lds(xn16 + ((size_t)b * NN + n0) * CC, xs, 16 * CC * 2, tid);
  __syncthreads();

  const int mrow = (lane >> 4) << 3;
  const int ncol = lane & 15;

  // tiles wave*12 .. wave*12+11 ; tile t -> mat = t>>5, ct = t&31
  for (int pp = 0; pp < 6; ++pp) {
    const int t0 = wave * 12 + pp * 2;
    const int t1 = t0 + 1;
    const int mat0 = t0 >> 5, ct0 = t0 & 31;
    const int mat1 = t1 >> 5, ct1 = t1 & 31;
    const _Float16* wA0 = w16 + (size_t)mat0 * CC * CC + (size_t)(ct0 * 16) * CC;
    const _Float16* wA1 = w16 + (size_t)mat1 * CC * CC + (size_t)(ct1 * 16) * CC;
    v8f acc0 = {}, acc1 = {};
#pragma unroll 4
    for (int kc = 0; kc < CC; kc += 32) {
      v16h bm = load_b32x16(xs + kc, CC, lane);       // LDS, shared by pair
      v16h a0 = load_a16x32(wA0 + kc, CC, lane);
      v16h a1 = load_a16x32(wA1 + kc, CC, lane);
      acc0 = wmma_f16(a0, bm, acc0);
      acc1 = wmma_f16(a1, bm, acc1);
    }
    qkv_store(mat0, ct0 * 16, acc0, bq, bk, bv, q16, k16, v16, b, n0, mrow, ncol);
    qkv_store(mat1, ct1 * 16, acc1, bq, bk, bv, q16, k16, v16, b, n0, mrow, ncol);
  }
}

// ---------------------------------------------------------------------------
// 4) Attention: one workgroup per (b, 16-row i-tile). 8 waves.
//    q-tile staged to LDS once; phase 1 pairs j-tiles (A reused from LDS),
//    softmax in f32 math over a 32KB f16 LDS score buffer; phase 2 pairs
//    c-tiles (A = probabilities from LDS, B = v from global, K-contiguous).
// ---------------------------------------------------------------------------
__global__ void attn_kernel(const _Float16* __restrict__ q16,
                            const _Float16* __restrict__ k16,
                            const _Float16* __restrict__ v16,
                            _Float16* __restrict__ outT) {
  __shared__ _Float16 qs[16 * CC];     // 16 KB q^T tile [i_local][c]
  __shared__ _Float16 P16[16 * NN];    // 32 KB probabilities [i_local][j]
  __shared__ float    red[16 * 16];

  const int tid  = threadIdx.x;
  const int lane = tid & 31;
  const int wave = tid >> 5;
  const int b    = blockIdx.x >> 6;
  const int it   = blockIdx.x & 63;
  const int i0   = it * 16;
  const int mrow = (lane >> 4) << 3;
  const int ncol = lane & 15;
  const float scale = 0.044194173824159216f;   // 1/sqrt(512)

  stage_to_lds(q16 + ((size_t)b * NN + i0) * CC, qs, 16 * CC * 2, tid);
  __syncthreads();

  // ---- Phase 1: scores, 4 j-tile pairs per wave ----
  for (int pp = 0; pp < 4; ++pp) {
    const int jt0 = wave * 8 + pp * 2;
    const int jt1 = jt0 + 1;
    const _Float16* kb0 = k16 + ((size_t)b * NN + jt0 * 16) * CC;
    const _Float16* kb1 = k16 + ((size_t)b * NN + jt1 * 16) * CC;
    v8f acc0 = {}, acc1 = {};
#pragma unroll 4
    for (int kc = 0; kc < CC; kc += 32) {
      v16h a  = load_a16x32(qs + kc, CC, lane);       // LDS, shared by pair
      v16h b0 = load_b32x16(kb0 + kc, CC, lane);
      v16h b1 = load_b32x16(kb1 + kc, CC, lane);
      acc0 = wmma_f16(a, b0, acc0);
      acc1 = wmma_f16(a, b1, acc1);
    }
#pragma unroll
    for (int r = 0; r < 8; ++r) {
      P16[(mrow + r) * NN + jt0 * 16 + ncol] = (_Float16)(acc0[r] * scale);
      P16[(mrow + r) * NN + jt1 * 16 + ncol] = (_Float16)(acc1[r] * scale);
    }
  }
  __syncthreads();

  // ---- Softmax: 16 threads per row, 64 elements each (f32 math) ----
  {
    const int row   = tid >> 4;
    const int chunk = tid & 15;
    _Float16* pr = P16 + row * NN;
    float m = -3.0e38f;
    for (int j = chunk * 64; j < chunk * 64 + 64; ++j)
      m = fmaxf(m, (float)pr[j]);
    red[row * 16 + chunk] = m;
    __syncthreads();
    float rm = -3.0e38f;
    for (int t = 0; t < 16; ++t) rm = fmaxf(rm, red[row * 16 + t]);
    float s = 0.f;
    for (int j = chunk * 64; j < chunk * 64 + 64; ++j) {
      float e = __expf((float)pr[j] - rm);
      pr[j] = (_Float16)e;
      s += e;
    }
    __syncthreads();                  // all rm reads done before red reuse
    red[row * 16 + chunk] = s;
    __syncthreads();
    float tot = 0.f;
    for (int t = 0; t < 16; ++t) tot += red[row * 16 + t];
    const float inv = 1.f / tot;
    for (int j = chunk * 64; j < chunk * 64 + 64; ++j)
      pr[j] = (_Float16)((float)pr[j] * inv);
  }
  __syncthreads();

  // ---- Phase 2: out^T = P * v^T, 2 c-tile pairs per wave ----
  for (int pp = 0; pp < 2; ++pp) {
    const int ct0 = wave * 4 + pp * 2;
    const int ct1 = ct0 + 1;
    const _Float16* vb0 = v16 + ((size_t)b * CC + ct0 * 16) * NN;
    const _Float16* vb1 = v16 + ((size_t)b * CC + ct1 * 16) * NN;
    v8f acc0 = {}, acc1 = {};
#pragma unroll 2
    for (int kt = 0; kt < NN; kt += 32) {
      v16h a  = load_a16x32(P16 + kt, NN, lane);      // LDS, shared by pair
      v16h b0 = load_b32x16(vb0 + kt, NN, lane);
      v16h b1 = load_b32x16(vb1 + kt, NN, lane);
      acc0 = wmma_f16(a, b0, acc0);
      acc1 = wmma_f16(a, b1, acc1);
    }
#pragma unroll
    for (int r = 0; r < 8; ++r) {
      outT[((size_t)b * NN + i0 + mrow + r) * CC + ct0 * 16 + ncol] = (_Float16)acc0[r];
      outT[((size_t)b * NN + i0 + mrow + r) * CC + ct1 * 16 + ncol] = (_Float16)acc1[r];
    }
  }
}

// ---------------------------------------------------------------------------
// 5) Output projection + residual: out[b][c][n] = x + wp*attn_out + bp
//    Same LDS-staged structure as qkv (one block per (b, n-tile)).
// ---------------------------------------------------------------------------
__global__ void proj_kernel(const _Float16* __restrict__ outT,
                            const _Float16* __restrict__ wp16,
                            const float* __restrict__ bp,
                            const float* __restrict__ x,
                            float* __restrict__ out) {
  __shared__ _Float16 os[16 * CC];     // 16 KB attn-out tile [n_local][c]
  const int tid  = threadIdx.x;
  const int lane = tid & 31;
  const int wave = tid >> 5;
  const int b  = blockIdx.x >> 6;
  const int nt = blockIdx.x & 63;
  const int n0 = nt * 16;

  stage_to_lds(outT + ((size_t)b * NN + n0) * CC, os, 16 * CC * 2, tid);
  __syncthreads();

  const int mrow = (lane >> 4) << 3;
  const int ncol = lane & 15;

  for (int pp = 0; pp < 2; ++pp) {     // 32 cout tiles, 4/wave, paired
    const int ct0 = wave * 4 + pp * 2;
    const int ct1 = ct0 + 1;
    const _Float16* wA0 = wp16 + (size_t)(ct0 * 16) * CC;
    const _Float16* wA1 = wp16 + (size_t)(ct1 * 16) * CC;
    v8f acc0 = {}, acc1 = {};
#pragma unroll 4
    for (int kc = 0; kc < CC; kc += 32) {
      v16h bm = load_b32x16(os + kc, CC, lane);       // LDS, shared by pair
      v16h a0 = load_a16x32(wA0 + kc, CC, lane);
      v16h a1 = load_a16x32(wA1 + kc, CC, lane);
      acc0 = wmma_f16(a0, bm, acc0);
      acc1 = wmma_f16(a1, bm, acc1);
    }
#pragma unroll
    for (int r = 0; r < 8; ++r) {
      const int c0 = ct0 * 16 + mrow + r;
      const int c1 = ct1 * 16 + mrow + r;
      const size_t i0 = ((size_t)b * CC + c0) * NN + n0 + ncol;
      const size_t i1 = ((size_t)b * CC + c1) * NN + n0 + ncol;
      out[i0] = x[i0] + acc0[r] + bp[c0];
      out[i1] = x[i1] + acc1[r] + bp[c1];
    }
  }
}

// ---------------------------------------------------------------------------
extern "C" void kernel_launch(void* const* d_in, const int* in_sizes, int n_in,
                              void* d_out, int out_size, void* d_ws, size_t ws_size,
                              hipStream_t stream) {
  const float* x  = (const float*)d_in[0];
  const float* gs = (const float*)d_in[1];
  const float* gb = (const float*)d_in[2];
  const float* wq = (const float*)d_in[3];
  const float* bq = (const float*)d_in[4];
  const float* wk = (const float*)d_in[5];
  const float* bk = (const float*)d_in[6];
  const float* wv = (const float*)d_in[7];
  const float* bv = (const float*)d_in[8];
  const float* wp = (const float*)d_in[9];
  const float* bp = (const float*)d_in[10];
  float* out = (float*)d_out;

  char* ws = (char*)d_ws;
  _Float16* w16  = (_Float16*)ws;  ws += (size_t)4 * CC * CC * sizeof(_Float16);
  _Float16* xn16 = (_Float16*)ws;  ws += (size_t)BN * NN * CC * sizeof(_Float16);
  _Float16* q16  = (_Float16*)ws;  ws += (size_t)BN * NN * CC * sizeof(_Float16);
  _Float16* k16  = (_Float16*)ws;  ws += (size_t)BN * NN * CC * sizeof(_Float16);
  _Float16* v16  = (_Float16*)ws;  ws += (size_t)BN * NN * CC * sizeof(_Float16);
  _Float16* oT16 = (_Float16*)ws;  ws += (size_t)BN * NN * CC * sizeof(_Float16);

  gn_kernel  <<<BN * 32, 256, 0, stream>>>(x, gs, gb, xn16);
  wcvt_kernel<<<4 * CC * CC / 256, 256, 0, stream>>>(wq, wk, wv, wp, w16);
  qkv_kernel <<<BN * 64, 256, 0, stream>>>(xn16, w16, bq, bk, bv, q16, k16, v16);
  attn_kernel<<<BN * 64, 256, 0, stream>>>(q16, k16, v16, oT16);
  proj_kernel<<<BN * 64, 256, 0, stream>>>(oT16, w16 + (size_t)3 * CC * CC, bp, x, out);
}
